// AttentionLayer_25245817766159
// MI455X (gfx1250) — matrix operand order repaired and checked by reading, not compiled, not run on hardware
//
#include <hip/hip_runtime.h>

// Problem sizes (fixed by the reference)
#define BB 16
#define SS 2048
#define DD 512
#define HH 64

typedef __attribute__((ext_vector_type(16))) __bf16 v16bf;
typedef __attribute__((ext_vector_type(8)))  float  v8f;

union Frag16 { v16bf v; unsigned u[8]; };

// round-to-nearest-even f32 -> bf16 (as raw u16)
__device__ __forceinline__ unsigned short f2bf(float x) {
  unsigned u = __float_as_uint(x);
  u += 0x7FFFu + ((u >> 16) & 1u);
  return (unsigned short)(u >> 16);
}

// ---------------------------------------------------------------------------
// Kernel 0: zero the output accumulator (d_out is atomically accumulated).
// ---------------------------------------------------------------------------
__global__ void zero_out_kernel(float* __restrict__ out, int n) {
  int i = blockIdx.x * blockDim.x + threadIdx.x;
  if (i < n) out[i] = 0.0f;
}

// ---------------------------------------------------------------------------
// Kernel 1: QKV projection  [S,D]fp32 @ [D,H]fp32 -> [S,H]bf16 (per batch).
// grid = (S/128, B, 3 {Q,K,V}), block = 256 (8 waves, 16 rows each).
// K-loop over D in steps of 32 with v_wmma_f32_16x16x32_bf16.
// Q is pre-scaled by 1/sqrt(H) = 0.125 so the attention kernel skips it.
// V is stored TRANSPOSED per 32-key chunk: Vt[b][chunk][h][t_local]
// (contiguous 4 KB per chunk) so the attention kernel can DMA it into LDS
// with global_load_async_to_lds_b128 — no in-flight transpose needed.
// ---------------------------------------------------------------------------
__global__ __launch_bounds__(256)
void qkv_proj_kernel(const float* __restrict__ X,
                     const float* __restrict__ Wq, const float* __restrict__ bq,
                     const float* __restrict__ Wk, const float* __restrict__ bk,
                     const float* __restrict__ Wv, const float* __restrict__ bv,
                     unsigned short* __restrict__ Qo,
                     unsigned short* __restrict__ Ko,
                     unsigned short* __restrict__ Vo) {
  const int mat = blockIdx.z;
  const float* W    = (mat == 0) ? Wq : (mat == 1) ? Wk : Wv;
  const float* bias = (mat == 0) ? bq : (mat == 1) ? bk : bv;
  unsigned short* out = (mat == 0) ? Qo : (mat == 1) ? Ko : Vo;
  const float scale = (mat == 0) ? 0.125f : 1.0f;  // fold 1/sqrt(H) into Q

  const int b    = blockIdx.y;
  const int row0 = blockIdx.x * 128;
  const int tid  = threadIdx.x;
  const int lane = tid & 31, wave = tid >> 5;
  const int half = lane >> 4, mn = lane & 15;

  __shared__ unsigned short Atile[128 * 32];  // X chunk, bf16, row-major (k contiguous)
  __shared__ unsigned short Btile[64 * 32];   // W chunk transposed: Bt[h][k], k contiguous

  v8f acc[4];
#pragma unroll
  for (int j = 0; j < 4; ++j)
#pragma unroll
    for (int e = 0; e < 8; ++e) acc[j][e] = 0.0f;

  const float* Xrow = X + ((size_t)b * SS + row0) * DD;

  for (int k0 = 0; k0 < DD; k0 += 32) {
    __syncthreads();  // protect LDS from readers of the previous step
    // stage A: 128x32 fp32 -> bf16 (coalesced: consecutive tid = consecutive k)
    for (int i = tid; i < 128 * 32; i += 256) {
      int r = i >> 5, c = i & 31;
      Atile[i] = f2bf(Xrow[(size_t)r * DD + k0 + c]);
    }
    // stage B transposed: W[k0+k][h] -> Bt[h*32+k] (coalesced along h)
    for (int i = tid; i < 64 * 32; i += 256) {
      int h = i & 63, k = i >> 6;
      Btile[h * 32 + k] = f2bf(W[(size_t)(k0 + k) * HH + h]);
    }
    __syncthreads();

    // A fragment: lane=row, regs pack K per ISA A-layout (8-interleaved)
    Frag16 a;
    {
      const unsigned short* p = Atile + (wave * 16 + mn) * 32 + half * 8;
#pragma unroll
      for (int r = 0; r < 8; ++r)
        a.u[r] = *(const unsigned*)(p + ((r >> 2) << 4) + ((r & 3) << 1));
    }
#pragma unroll
    for (int j = 0; j < 4; ++j) {
      // B fragment: lane=col(h), K 8-interleaved across lane halves (same as A)
      Frag16 bf;
      {
        const unsigned short* p = Btile + (j * 16 + mn) * 32 + half * 8;
#pragma unroll
        for (int r = 0; r < 8; ++r) bf.u[r] = *(const unsigned*)(p + ((r >> 2) << 4) + ((r & 3) << 1));
      }
      acc[j] = __builtin_amdgcn_wmma_f32_16x16x32_bf16(
          false, a.v, false, bf.v, (short)0, acc[j], false, false);
    }
  }

  // epilogue: bias, scale, bf16 store. C layout: lane=col, reg r = row r+8*half
#pragma unroll
  for (int j = 0; j < 4; ++j) {
    int h = j * 16 + mn;
    float bb = bias[h];
#pragma unroll
    for (int r = 0; r < 8; ++r) {
      int m   = r + 8 * half;
      int row = row0 + wave * 16 + m;
      unsigned short val = f2bf((acc[j][r] + bb) * scale);
      if (mat == 2) {
        // transposed chunked layout: [b][row/32][h][row%32]
        size_t idx = (((size_t)b * (SS / 32) + (row >> 5)) * HH + h) * 32 + (row & 31);
        out[idx] = val;
      } else {
        out[((size_t)b * SS + row) * HH + h] = val;
      }
    }
  }
}

// ---------------------------------------------------------------------------
// Kernel 2: flash attention + mean pool.
// grid = (S/64, B), block = 128 (4 waves, 16 query rows each).
// Per key chunk of 32: V tile DMA'd into LDS with global_load_async_to_lds_b128
// (overlapped with the score WMMAs, retired via s_wait_asynccnt), scores =
// Q(16x64) @ K^T (4 WMMAs), online softmax (f32, wave32 shuffles),
// context += P(16x32) @ V(32x64) (4 WMMAs). K rows for the next chunk are
// prefetched (global_prefetch_b8).
// ---------------------------------------------------------------------------
__global__ __launch_bounds__(128)
void attn_kernel(const unsigned short* __restrict__ Q,
                 const unsigned short* __restrict__ K,
                 const unsigned short* __restrict__ Vt_ws,
                 float* __restrict__ out) {
  const int b    = blockIdx.y;
  const int s0   = blockIdx.x * 64;
  const int tid  = threadIdx.x;
  const int lane = tid & 31, wave = tid >> 5;
  const int half = lane >> 4, mn = lane & 15;

  __shared__ unsigned short Plds[4 * 16 * 32];  // per-wave P transpose buffer
  __shared__ unsigned short Vt[64 * 32];        // V chunk (already transposed): Vt[h][t]

  // raw LDS byte offset of Vt for the async-DMA destination (ISA 10.2:
  // the LDS address is the low 32 bits of the generic address)
  const unsigned vt_lds_base = (unsigned)(size_t)(&Vt[0]);

  // Q fragments for this wave's 16 rows: K(=h) dim 64 -> 2 A-frags, kept in regs
  Frag16 qa[2];
  {
    const unsigned short* qrow = Q + ((size_t)b * SS + s0 + wave * 16 + mn) * HH;
#pragma unroll
    for (int qi = 0; qi < 2; ++qi) {
      const unsigned short* p = qrow + qi * 32 + half * 8;
#pragma unroll
      for (int r = 0; r < 8; ++r)
        qa[qi].u[r] = *(const unsigned*)(p + ((r >> 2) << 4) + ((r & 3) << 1));
    }
  }

  float m_i[8], l_i[8];
  v8f ctx[4];
#pragma unroll
  for (int r = 0; r < 8; ++r) { m_i[r] = -3.0e38f; l_i[r] = 0.0f; }
#pragma unroll
  for (int j = 0; j < 4; ++j)
#pragma unroll
    for (int e = 0; e < 8; ++e) ctx[j][e] = 0.0f;

  for (int t0 = 0; t0 < SS; t0 += 32) {
    __syncthreads();  // previous iteration's Vt readers are done

    // async DMA: 4 KB contiguous V chunk -> LDS (2 x b128 per thread)
    {
      const unsigned short* vch =
          Vt_ws + (((size_t)b * (SS / 32) + (t0 >> 5)) * (HH * 32));
      unsigned long long ga = (unsigned long long)(size_t)(vch) + (size_t)tid * 16u;
      unsigned la = vt_lds_base + (unsigned)tid * 16u;
      asm volatile("global_load_async_to_lds_b128 %0, %1, off"
                   :: "v"(la), "v"(ga) : "memory");
      asm volatile("global_load_async_to_lds_b128 %0, %1, off offset:2048"
                   :: "v"(la), "v"(ga) : "memory");
    }

    // prefetch next chunk's K rows into cache while we compute
    if (t0 + 32 < SS)
      __builtin_prefetch(&K[((size_t)b * SS + t0 + 32 + lane) * HH], 0, 3);

    // scores: 2 key tiles x 2 h-steps. K read straight from global (L2-resident).
    v8f sc[2];
#pragma unroll
    for (int tt = 0; tt < 2; ++tt)
#pragma unroll
      for (int e = 0; e < 8; ++e) sc[tt][e] = 0.0f;
#pragma unroll
    for (int tt = 0; tt < 2; ++tt) {
      const unsigned short* krow =
          K + ((size_t)b * SS + t0 + tt * 16 + mn) * HH;
#pragma unroll
      for (int hs = 0; hs < 2; ++hs) {
        Frag16 kb;
        const unsigned short* p = krow + hs * 32 + half * 8;
#pragma unroll
        for (int r = 0; r < 8; ++r) kb.u[r] = *(const unsigned*)(p + ((r >> 2) << 4) + ((r & 3) << 1));
        sc[tt] = __builtin_amdgcn_wmma_f32_16x16x32_bf16(
            false, qa[hs].v, false, kb.v, (short)0, sc[tt], false, false);
      }
    }

    // online softmax over this 32-key chunk (row r+8*half lives in reg r)
#pragma unroll
    for (int r = 0; r < 8; ++r) {
      float mx = fmaxf(sc[0][r], sc[1][r]);
#pragma unroll
      for (int msk = 8; msk >= 1; msk >>= 1)
        mx = fmaxf(mx, __shfl_xor(mx, msk, 32));
      float mnew  = fmaxf(m_i[r], mx);
      float alpha = __expf(m_i[r] - mnew);
      m_i[r] = mnew;
      float p0 = __expf(sc[0][r] - mnew);
      float p1 = __expf(sc[1][r] - mnew);
      float rs = p0 + p1;
#pragma unroll
      for (int msk = 8; msk >= 1; msk >>= 1)
        rs += __shfl_xor(rs, msk, 32);
      l_i[r] = l_i[r] * alpha + rs;
#pragma unroll
      for (int j = 0; j < 4; ++j) ctx[j][r] *= alpha;
      int m = r + 8 * half;  // C-layout row -> P row-major in LDS
      Plds[wave * 512 + m * 32 + mn]      = f2bf(p0);
      Plds[wave * 512 + m * 32 + 16 + mn] = f2bf(p1);
    }

    // retire this wave's async DMA before the barrier; compiler waits dscnt for
    // the ds_store's on its own.
    asm volatile("s_wait_asynccnt 0x0" ::: "memory");
    __syncthreads();  // Vt DMA complete block-wide; P committed

    // P as A-fragment (16 rows x K=32 keys) from LDS
    Frag16 pa;
    {
      const unsigned short* p = Plds + wave * 512 + mn * 32 + half * 8;
#pragma unroll
      for (int r = 0; r < 8; ++r)
        pa.u[r] = *(const unsigned*)(p + ((r >> 2) << 4) + ((r & 3) << 1));
    }
    // context += P @ V  (4 N-tiles over h)
#pragma unroll
    for (int j = 0; j < 4; ++j) {
      Frag16 vb;
      const unsigned short* p = Vt + (j * 16 + mn) * 32 + half * 8;
#pragma unroll
      for (int r = 0; r < 8; ++r) vb.u[r] = *(const unsigned*)(p + ((r >> 2) << 4) + ((r & 3) << 1));
      ctx[j] = __builtin_amdgcn_wmma_f32_16x16x32_bf16(
          false, pa.v, false, vb.v, (short)0, ctx[j], false, false);
    }
  }

  // finalize softmax and mean-pool this wave's 16 rows into out[b][h]
  float inv[8];
#pragma unroll
  for (int r = 0; r < 8; ++r) inv[r] = 1.0f / l_i[r];
#pragma unroll
  for (int j = 0; j < 4; ++j) {
    float s = 0.0f;
#pragma unroll
    for (int r = 0; r < 8; ++r) s += ctx[j][r] * inv[r];
    s += __shfl_xor(s, 16, 32);  // add the other half's 8 rows
    if (lane < 16)
      atomicAdd(&out[b * HH + j * 16 + mn], s * (1.0f / (float)SS));
  }
}

// ---------------------------------------------------------------------------
// Host-side launcher. Workspace layout (bf16 as u16):
//   Q : [B,S,H]                 @ 0      (4 MB)
//   K : [B,S,H]                 @ 4 MB
//   Vt: [B,S/32,H,32] (chunked) @ 8 MB   (12 MB total)
// ---------------------------------------------------------------------------
extern "C" void kernel_launch(void* const* d_in, const int* in_sizes, int n_in,
                              void* d_out, int out_size, void* d_ws, size_t ws_size,
                              hipStream_t stream) {
  (void)in_sizes; (void)n_in; (void)out_size; (void)ws_size;
  const float* X  = (const float*)d_in[0];
  const float* Wq = (const float*)d_in[1];
  const float* bq = (const float*)d_in[2];
  const float* Wk = (const float*)d_in[3];
  const float* bk = (const float*)d_in[4];
  const float* Wv = (const float*)d_in[5];
  const float* bv = (const float*)d_in[6];
  float* out = (float*)d_out;

  unsigned short* Qw = (unsigned short*)d_ws;
  unsigned short* Kw = Qw + (size_t)BB * SS * HH;
  unsigned short* Vw = Kw + (size_t)BB * SS * HH;

  zero_out_kernel<<<dim3((BB * HH + 255) / 256), dim3(256), 0, stream>>>(out, BB * HH);
  qkv_proj_kernel<<<dim3(SS / 128, BB, 3), dim3(256), 0, stream>>>(
      X, Wq, bq, Wk, bk, Wv, bv, Qw, Kw, Vw);
  attn_kernel<<<dim3(SS / 64, BB), dim3(128), 0, stream>>>(Qw, Kw, Vw, out);
}
